// HeteroGraphSAGEEncoder_27754078667401
// MI455X (gfx1250) — matrix operand order repaired and checked by reading, not compiled, not run on hardware
//
#include <hip/hip_runtime.h>
#include <hip/hip_bf16.h>

// ---------------- problem constants ----------------
#define S0   200000
#define D0   40000
#define D1   8192
#define E0   640000
#define E1   131072
#define DIN  256
#define DH   512
#define DOUT 256

// ---------------- types ----------------
typedef __attribute__((ext_vector_type(16))) __bf16 v16bf;
typedef __attribute__((ext_vector_type(8)))  __bf16 bf16x8;
typedef __attribute__((ext_vector_type(4)))  __bf16 bf16x4;
typedef __attribute__((ext_vector_type(8)))  float  v8f;

__device__ __forceinline__ __bf16 f2bf(float f) {
    unsigned u = __builtin_bit_cast(unsigned, f);
    u += 0x7FFFu + ((u >> 16) & 1u);          // round to nearest even
    unsigned short h = (unsigned short)(u >> 16);
    return __builtin_bit_cast(__bf16, h);
}

__device__ __forceinline__ void atomic_add_f32(float* p, float v) {
    __hip_atomic_fetch_add(p, v, __ATOMIC_RELAXED, __HIP_MEMORY_SCOPE_AGENT);
}

// ---------------- utility kernels ----------------
__global__ __launch_bounds__(256)
void fill_zero_f32(float* __restrict__ p, int n4) {
    int i = blockIdx.x * blockDim.x + threadIdx.x;
    if (i < n4) ((float4*)p)[i] = make_float4(0.f, 0.f, 0.f, 0.f);
}

__global__ __launch_bounds__(256)
void f32_to_bf16_k(const float* __restrict__ src, __bf16* __restrict__ dst, int n4) {
    int i = blockIdx.x * blockDim.x + threadIdx.x;
    if (i < n4) {
        float4 v = ((const float4*)src)[i];
        bf16x4 o; o[0] = f2bf(v.x); o[1] = f2bf(v.y); o[2] = f2bf(v.z); o[3] = f2bf(v.w);
        ((bf16x4*)dst)[i] = o;
    }
}

// Wt[n,k] = bf16(W[k,n]); W row-major [K,N]
__global__ __launch_bounds__(256)
void f32_to_bf16_transpose(const float* __restrict__ W, __bf16* __restrict__ Wt,
                           int K, int N) {
    int tid = blockIdx.x * blockDim.x + threadIdx.x;
    if (tid >= N * K) return;
    int n = tid / K, k = tid - n * K;
    Wt[(size_t)n * K + k] = f2bf(W[(size_t)k * N + n]);
}

// edge-parallel gather + scatter-add (mean numerator) and degree count
__global__ __launch_bounds__(256)
void scatter_agg(const float* __restrict__ xsrc, const int* __restrict__ esrc,
                 const int* __restrict__ edst, float* __restrict__ agg,
                 float* __restrict__ deg, int E, int D) {
    int chunks = D >> 2;
    int tid = blockIdx.x * blockDim.x + threadIdx.x;
    if (tid >= E * chunks) return;
    int e = tid / chunks;
    int c = (tid - e * chunks) << 2;
    int s = esrc[e], d = edst[e];
    float4 v = *(const float4*)(xsrc + (size_t)s * D + c);
    float* p = agg + (size_t)d * D + c;
    atomic_add_f32(p + 0, v.x);
    atomic_add_f32(p + 1, v.y);
    atomic_add_f32(p + 2, v.z);
    atomic_add_f32(p + 3, v.w);
    if (c == 0) atomic_add_f32(deg + d, 1.0f);
}

// h = agg / max(deg,1), converted to bf16
__global__ __launch_bounds__(256)
void norm_to_bf16(const float* __restrict__ agg, const float* __restrict__ deg,
                  __bf16* __restrict__ out, int rows, int D) {
    int chunks = D >> 2;
    int tid = blockIdx.x * blockDim.x + threadIdx.x;
    if (tid >= rows * chunks) return;
    int r = tid / chunks;
    int c = (tid - r * chunks) << 2;
    float inv = 1.0f / fmaxf(deg[r], 1.0f);
    float4 v = *(const float4*)(agg + (size_t)r * D + c);
    bf16x4 o; o[0] = f2bf(v.x * inv); o[1] = f2bf(v.y * inv);
              o[2] = f2bf(v.z * inv); o[3] = f2bf(v.w * inv);
    *(bf16x4*)(out + (size_t)r * D + c) = o;
}

// ---------------- WMMA GEMM: C = A0*B0^T + A1*B1^T + bias (opt relu) ----------------
// A row-major [M,K] bf16. Bt row-major [N,K] bf16 (pre-transposed weights).
// Block: 128 threads = 4 waves. Block tile 64(M) x 128(N); wave (wr,wc) -> 32x64.
// Tiles staged to LDS with CDNA5 async global->LDS copies (ASYNCcnt).
#define LDSL 40   // padded LDS row stride in bf16 elements (32 data + 8 pad)

__global__ __launch_bounds__(128)
void wmma_gemm_dual(const __bf16* __restrict__ A0, const __bf16* __restrict__ Bt0,
                    const __bf16* __restrict__ A1, const __bf16* __restrict__ Bt1,
                    const float* __restrict__ bias, float* __restrict__ C,
                    int M, int N, int K, int relu) {
    // A at rows [0,64), B at rows [64,192)
    __shared__ __align__(16) __bf16 lds[(64 + 128) * LDSL];
    __bf16* As = lds;
    __bf16* Bs = lds + 64 * LDSL;

    // Low 32 bits of a flat shared pointer == LDS byte address.
    // The ptrtoint also captures `lds`, so the asm "memory" clobber aliases it and the
    // fragment ds_loads below cannot be folded away.
    const unsigned lds_base = (unsigned)(uintptr_t)(void*)lds;

    const int tid  = threadIdx.x;
    const int lane = tid & 31;
    const int wave = tid >> 5;       // 0..3
    const int wr   = wave >> 1;      // row half  (0..1) -> 32 rows
    const int wc   = wave & 1;       // col half  (0..1) -> 64 cols
    const int half = lane >> 4;      // 0/1
    const int mrow = lane & 15;

    const int blockN = blockIdx.x * 128;
    const int blockM = blockIdx.y * 64;

    v8f acc[2][4] = {};

    #pragma unroll
    for (int seg = 0; seg < 2; ++seg) {
        const __bf16* __restrict__ A = seg ? A1 : A0;
        const __bf16* __restrict__ B = seg ? Bt1 : Bt0;
        for (int kb = 0; kb < K; kb += 32) {
            // ---- async stage A tile 64x32 (2 chunks of 8 bf16 per thread) ----
            #pragma unroll
            for (int cc = 0; cc < 2; ++cc) {
                int c    = tid + cc * 128;          // 0..255
                int r    = c >> 2;                  // 0..63
                int col8 = (c & 3) * 8;             // 0,8,16,24
                unsigned lo = lds_base + (unsigned)((r * LDSL + col8) * 2);
                unsigned long long ga =
                    (unsigned long long)(A + (size_t)(blockM + r) * K + kb + col8);
                asm volatile("global_load_async_to_lds_b128 %0, %1, off"
                             :: "v"(lo), "v"(ga) : "memory");
            }
            // ---- async stage Bt tile 128x32 (4 chunks per thread) ----
            #pragma unroll
            for (int cc = 0; cc < 4; ++cc) {
                int c    = tid + cc * 128;          // 0..511
                int n    = c >> 2;                  // 0..127
                int col8 = (c & 3) * 8;
                unsigned lo = lds_base + (unsigned)(((64 + n) * LDSL + col8) * 2);
                unsigned long long ga =
                    (unsigned long long)(B + (size_t)(blockN + n) * K + kb + col8);
                asm volatile("global_load_async_to_lds_b128 %0, %1, off"
                             :: "v"(lo), "v"(ga) : "memory");
            }
#if __has_builtin(__builtin_amdgcn_s_wait_asynccnt)
            __builtin_amdgcn_s_wait_asynccnt(0);
#else
            asm volatile("s_wait_asynccnt 0x0" ::: "memory");
#endif
            __syncthreads();

            // ---- A fragments per ISA layout:
            // lanes 0-15: row=lane, K {0-7,16-23}; lanes 16-31: row=lane-16, K {8-15,24-31}
            v16bf afr[2];
            #pragma unroll
            for (int a = 0; a < 2; ++a) {
                int arow = wr * 32 + a * 16 + mrow;
                bf16x8 lo8 = *(const bf16x8*)(As + arow * LDSL + 8 * half);
                bf16x8 hi8 = *(const bf16x8*)(As + arow * LDSL + 16 + 8 * half);
                #pragma unroll
                for (int j = 0; j < 8; ++j) { afr[a][j] = lo8[j]; afr[a][8 + j] = hi8[j]; }
            }
            // ---- B fragments: lanes 0-15 col=lane K {0-7,16-23}; lanes 16-31 col=lane-16 K {8-15,24-31}
            v16bf bfr[4];
            #pragma unroll
            for (int t = 0; t < 4; ++t) {
                int n = wc * 64 + t * 16 + mrow;
                bf16x8 lo8 = *(const bf16x8*)(Bs + n * LDSL + 8 * half);
                bf16x8 hi8 = *(const bf16x8*)(Bs + n * LDSL + 16 + 8 * half);
                #pragma unroll
                for (int j = 0; j < 8; ++j) { bfr[t][j] = lo8[j]; bfr[t][8 + j] = hi8[j]; }
            }
            // ---- 8 WMMAs per wave per K-step ----
            #pragma unroll
            for (int a = 0; a < 2; ++a)
                #pragma unroll
                for (int t = 0; t < 4; ++t)
                    acc[a][t] = __builtin_amdgcn_wmma_f32_16x16x32_bf16(
                        false, afr[a], false, bfr[t], (short)0, acc[a][t], false, false);
            __syncthreads();
        }
    }

    // ---- epilogue: C layout: VGPR v -> row = v + 8*half, col = lane&15 ----
    #pragma unroll
    for (int a = 0; a < 2; ++a) {
        #pragma unroll
        for (int t = 0; t < 4; ++t) {
            int ncol = blockN + wc * 64 + t * 16 + mrow;
            float bb = bias[ncol];
            #pragma unroll
            for (int v = 0; v < 8; ++v) {
                int row = blockM + wr * 32 + a * 16 + v + 8 * half;
                float val = acc[a][t][v] + bb;
                if (relu) val = fmaxf(val, 0.0f);
                C[(size_t)row * N + ncol] = val;
            }
        }
    }
}

// ---------------- host driver ----------------
extern "C" void kernel_launch(void* const* d_in, const int* in_sizes, int n_in,
                              void* d_out, int out_size, void* d_ws, size_t ws_size,
                              hipStream_t stream) {
    const float* x_user    = (const float*)d_in[0];
    const float* x_movie   = (const float*)d_in[1];
    const float* W1r_neigh = (const float*)d_in[2];
    const float* W1r_self  = (const float*)d_in[3];
    const float* b1r       = (const float*)d_in[4];
    const float* W1b_neigh = (const float*)d_in[5];
    const float* W1b_self  = (const float*)d_in[6];
    const float* b1b       = (const float*)d_in[7];
    const float* W2r_neigh = (const float*)d_in[8];
    const float* W2r_self  = (const float*)d_in[9];
    const float* b2r       = (const float*)d_in[10];
    const float* W2b_neigh = (const float*)d_in[11];
    const float* W2b_self  = (const float*)d_in[12];
    const float* b2b       = (const float*)d_in[13];
    const int* e0r_src = (const int*)d_in[14];
    const int* e0r_dst = (const int*)d_in[15];
    const int* e0b_src = (const int*)d_in[16];
    const int* e0b_dst = (const int*)d_in[17];
    const int* e1r_src = (const int*)d_in[18];
    const int* e1r_dst = (const int*)d_in[19];
    const int* e1b_src = (const int*)d_in[20];
    const int* e1b_dst = (const int*)d_in[21];

    // ---- workspace carve (256B aligned) ----
    char* p = (char*)d_ws;
    auto take = [&](size_t bytes) { char* r = p; p += (bytes + 255) & ~(size_t)255; return r; };
    float*  aggR = (float*)take((size_t)D0 * DIN * 4);   // reused for layer-2 agg (8192x512)
    float*  aggB = (float*)take((size_t)D0 * DIN * 4);
    float*  degR = (float*)take((size_t)D0 * 4);
    float*  degB = (float*)take((size_t)D0 * 4);
    __bf16* hR   = (__bf16*)take((size_t)D0 * DIN * 2);
    __bf16* hB   = (__bf16*)take((size_t)D0 * DIN * 2);
    __bf16* xmb  = (__bf16*)take((size_t)D0 * DIN * 2);
    __bf16* xub  = (__bf16*)take((size_t)D0 * DIN * 2);
    float*  m1   = (float*)take((size_t)D0 * DH * 4);
    float*  u1   = (float*)take((size_t)D0 * DH * 4);
    __bf16* h2R  = (__bf16*)take((size_t)D1 * DH * 2);
    __bf16* h2B  = (__bf16*)take((size_t)D1 * DH * 2);
    __bf16* m1s  = (__bf16*)take((size_t)D1 * DH * 2);
    __bf16* u1s  = (__bf16*)take((size_t)D1 * DH * 2);
    __bf16* Wt[8];
    for (int i = 0; i < 8; ++i) Wt[i] = (__bf16*)take((size_t)DIN * DH * 2); // 131072 elems each

    auto blocks = [](long long n, int tpb) { return dim3((unsigned)((n + tpb - 1) / tpb)); };

    // ======== Layer 1 ========
    fill_zero_f32<<<blocks((long long)D0 * DIN / 4, 256), 256, 0, stream>>>(aggR, D0 * DIN / 4);
    fill_zero_f32<<<blocks((long long)D0 * DIN / 4, 256), 256, 0, stream>>>(aggB, D0 * DIN / 4);
    fill_zero_f32<<<blocks(D0 / 4, 256), 256, 0, stream>>>(degR, D0 / 4);
    fill_zero_f32<<<blocks(D0 / 4, 256), 256, 0, stream>>>(degB, D0 / 4);

    scatter_agg<<<blocks((long long)E0 * (DIN / 4), 256), 256, 0, stream>>>(x_user,  e0r_src, e0r_dst, aggR, degR, E0, DIN);
    scatter_agg<<<blocks((long long)E0 * (DIN / 4), 256), 256, 0, stream>>>(x_movie, e0b_src, e0b_dst, aggB, degB, E0, DIN);

    norm_to_bf16<<<blocks((long long)D0 * (DIN / 4), 256), 256, 0, stream>>>(aggR, degR, hR, D0, DIN);
    norm_to_bf16<<<blocks((long long)D0 * (DIN / 4), 256), 256, 0, stream>>>(aggB, degB, hB, D0, DIN);

    f32_to_bf16_k<<<blocks((long long)D0 * DIN / 4, 256), 256, 0, stream>>>(x_movie, xmb, D0 * DIN / 4);
    f32_to_bf16_k<<<blocks((long long)D0 * DIN / 4, 256), 256, 0, stream>>>(x_user,  xub, D0 * DIN / 4);

    // transposed bf16 weights: layer1 [K=256,N=512] -> [512,256]; layer2 [512,256] -> [256,512]
    const float* wsrc[8] = {W1r_self, W1r_neigh, W1b_self, W1b_neigh,
                            W2r_self, W2r_neigh, W2b_self, W2b_neigh};
    const int wK[8] = {DIN, DIN, DIN, DIN, DH, DH, DH, DH};
    const int wN[8] = {DH, DH, DH, DH, DOUT, DOUT, DOUT, DOUT};
    for (int i = 0; i < 8; ++i)
        f32_to_bf16_transpose<<<blocks(DIN * DH, 256), 256, 0, stream>>>(wsrc[i], Wt[i], wK[i], wN[i]);

    // m1 = relu(x_movie[:D0]@W1r_self + hR@W1r_neigh + b1r); u1 analog
    wmma_gemm_dual<<<dim3(DH / 128, D0 / 64), 128, 0, stream>>>(xmb, Wt[0], hR, Wt[1], b1r, m1, D0, DH, DIN, 1);
    wmma_gemm_dual<<<dim3(DH / 128, D0 / 64), 128, 0, stream>>>(xub, Wt[2], hB, Wt[3], b1b, u1, D0, DH, DIN, 1);

    // ======== Layer 2 ========
    fill_zero_f32<<<blocks((long long)D1 * DH / 4, 256), 256, 0, stream>>>(aggR, D1 * DH / 4);
    fill_zero_f32<<<blocks((long long)D1 * DH / 4, 256), 256, 0, stream>>>(aggB, D1 * DH / 4);
    fill_zero_f32<<<blocks(D1 / 4, 256), 256, 0, stream>>>(degR, D1 / 4);
    fill_zero_f32<<<blocks(D1 / 4, 256), 256, 0, stream>>>(degB, D1 / 4);

    scatter_agg<<<blocks((long long)E1 * (DH / 4), 256), 256, 0, stream>>>(u1, e1r_src, e1r_dst, aggR, degR, E1, DH);
    scatter_agg<<<blocks((long long)E1 * (DH / 4), 256), 256, 0, stream>>>(m1, e1b_src, e1b_dst, aggB, degB, E1, DH);

    norm_to_bf16<<<blocks((long long)D1 * (DH / 4), 256), 256, 0, stream>>>(aggR, degR, h2R, D1, DH);
    norm_to_bf16<<<blocks((long long)D1 * (DH / 4), 256), 256, 0, stream>>>(aggB, degB, h2B, D1, DH);

    f32_to_bf16_k<<<blocks((long long)D1 * DH / 4, 256), 256, 0, stream>>>(m1, m1s, D1 * DH / 4);
    f32_to_bf16_k<<<blocks((long long)D1 * DH / 4, 256), 256, 0, stream>>>(u1, u1s, D1 * DH / 4);

    float* u2 = (float*)d_out;                       // first output
    float* m2 = (float*)d_out + (size_t)D1 * DOUT;   // second output

    // u2 = u1[:D1]@W2b_self + h2B@W2b_neigh + b2b
    wmma_gemm_dual<<<dim3(DOUT / 128, D1 / 64), 128, 0, stream>>>(u1s, Wt[6], h2B, Wt[7], b2b, u2, D1, DOUT, DH, 0);
    // m2 = m1[:D1]@W2r_self + h2R@W2r_neigh + b2r
    wmma_gemm_dual<<<dim3(DOUT / 128, D1 / 64), 128, 0, stream>>>(m1s, Wt[4], h2R, Wt[5], b2r, m2, D1, DOUT, DH, 0);
}
